// GNNMeshEmbed_49744311222713
// MI455X (gfx1250) — matrix operand order repaired and checked by reading, not compiled, not run on hardware
//
#include <hip/hip_runtime.h>
#include <hip/hip_bf16.h>

#define NN 100000
#define EE 1600000
#define HH 32
#define KK 25000
#define SS 131072   // next pow2 >= NN for bitonic sort

typedef unsigned long long u64;
typedef __attribute__((ext_vector_type(16))) _Float16 v16h;
typedef __attribute__((ext_vector_type(8)))  _Float16 v8h;
typedef __attribute__((ext_vector_type(8)))  float    v8f;
typedef __attribute__((ext_vector_type(4)))  unsigned u32x4;
typedef __attribute__((ext_vector_type(8)))  unsigned u32x8;

__device__ __forceinline__ v8f wmma_f16(v16h a, v16h b, v8f c) {
    return __builtin_amdgcn_wmma_f32_16x16x32_f16(
        false, a, false, b, (short)0, c, false, false);
}

__device__ __forceinline__ v16h cat8(v8h lo, v8h hi) {
    return __builtin_shufflevector(lo, hi, 0, 1, 2, 3, 4, 5, 6, 7,
                                   8, 9, 10, 11, 12, 13, 14, 15);
}

__device__ __forceinline__ float silu(float x) {
    return x / (1.0f + __expf(-x));   // native v_exp_f32 path
}

// ---------------------------------------------------------------------------
// Pre-pack Wm / Wu / Wo into per-lane WMMA B-fragment layout (f16).
// Fragment f = chunk*2 + tile; lane: n = lane&15, half = lane>>4;
// element e: W[(chunk*32 + (e < 8 ? half*8 + e : 16 + half*8 + e - 8))*32 + tile*16 + n].
// ---------------------------------------------------------------------------
__global__ void prep_frags_kernel(const float* __restrict__ Wm,
                                  const float* __restrict__ Wu,
                                  const float* __restrict__ Wo,
                                  _Float16* __restrict__ wmF,
                                  _Float16* __restrict__ wuF,
                                  _Float16* __restrict__ woF) {
    int t = threadIdx.x;          // 0..127
    int f = t >> 5, lane = t & 31;
    int chunk = f >> 1, tile = f & 1;
    int n = lane & 15, half = lane >> 4;
#pragma unroll
    for (int e = 0; e < 16; ++e) {
        int kk = (e < 8) ? (half * 8 + e) : (16 + half * 8 + (e - 8));
        int k = chunk * 32 + kk;
        wmF[(size_t)t * 16 + e] = (_Float16)Wm[k * HH + tile * 16 + n];
        wuF[(size_t)t * 16 + e] = (_Float16)Wu[k * HH + tile * 16 + n];
        if (chunk == 0)
            woF[(size_t)(tile * 32 + lane) * 16 + e] =
                (_Float16)Wo[kk * HH + tile * 16 + n];
    }
}

// ---------------------------------------------------------------------------
// h = x @ We + be + sincos_embed(pos); also write f16 mirror h16
// ---------------------------------------------------------------------------
__global__ void embed_kernel(const float* __restrict__ x,
                             const float* __restrict__ pos,
                             const float* __restrict__ We,
                             const float* __restrict__ be,
                             float* __restrict__ h,
                             _Float16* __restrict__ h16) {
    int t = blockIdx.x * blockDim.x + threadIdx.x;
    if (t >= NN * HH) return;
    int n = t >> 5;
    int f = t & 31;
    float acc = be[f];
    acc += x[n * 3 + 0] * We[0 * HH + f];
    acc += x[n * 3 + 1] * We[1 * HH + f];
    acc += x[n * 3 + 2] * We[2 * HH + f];
    int d = f >> 4;
    int j = f & 15;
    int jj = (j < 8) ? j : (j - 8);
    float omega = __expf(-(float)jj * (0.125f * 9.210340371976184f)); // 10000^(-jj/8)
    float v = pos[n * 2 + d] * omega;                                  // v in [0,1]
    acc += (j < 8) ? __sinf(v) : __cosf(v);
    h[t] = acc;
    h16[t] = (_Float16)acc;
}

__global__ void zero_f32(float* __restrict__ p, int n) {
    int i = blockIdx.x * blockDim.x + threadIdx.x;
    if (i < n) p[i] = 0.0f;
}

// ---------------------------------------------------------------------------
// Edge messages: msg = silu([h[dst], h[src], dist] @ Wm + bm); aggr[dst] += msg
// Wave handles 16 edges; 4x v_wmma_f32_16x16x32_f16.
// Wm fragments (4 KB) staged into LDS once per block via TDM tensor_load_to_lds.
// Grid is exactly EE/16/8 blocks -> every wave reaches __syncthreads().
// ---------------------------------------------------------------------------
__global__ void edge_msg_kernel(const _Float16* __restrict__ h16,
                                const float* __restrict__ pos,
                                const int*   __restrict__ edge,
                                const _Float16* __restrict__ wmF,
                                const float* __restrict__ Wm,
                                const float* __restrict__ bm,
                                float* __restrict__ aggr) {
    __shared__ _Float16 ldsWm[2048];          // 4 KB: 4 fragments x 32 lanes x 16 f16

    if (threadIdx.x < 32) {
        // Tensor DMA Descriptor (D#): 2D, data_size=4B, tile 1024x1 elements.
        unsigned lds = (unsigned)(size_t)(&ldsWm[0]);       // addr[31:0] = LDS byte addr
        unsigned long long ga = (unsigned long long)(size_t)wmF;
        u32x4 g0;
        g0[0] = 1u;                                          // count=1, user mode
        g0[1] = lds;                                         // lds_addr
        g0[2] = (unsigned)ga;                                // global_addr[31:0]
        g0[3] = ((unsigned)(ga >> 32) & 0x01FFFFFFu) | 0x80000000u; // ga[56:32] | type=2
        u32x8 g1;
        g1[0] = 0x00020000u;        // workgroup_mask=0, data_size=2 (4B), no flags
        g1[1] = 1024u << 16;        // tensor_dim0 = 1024
        g1[2] = 1u << 16;           // tensor_dim1 = 1
        g1[3] = 1024u << 16;        // tile_dim0 = 1024
        g1[4] = 1u;                 // tile_dim1 = 1, tile_dim2 = 0
        g1[5] = 1024u;              // tensor_dim0_stride = 1024
        g1[6] = 0u;
        g1[7] = 0u;
        asm volatile("tensor_load_to_lds %0, %1" :: "s"(g0), "s"(g1) : "memory");
        __builtin_amdgcn_s_wait_tensorcnt(0);
    }
    __syncthreads();

    int wave = (blockIdx.x * blockDim.x + threadIdx.x) >> 5;
    int lane = threadIdx.x & 31;
    int e0 = wave << 4;
    int half = lane >> 4;
    int m = lane & 15;
    int em = e0 + m;
    int src = edge[em * 2 + 0];
    int dst = edge[em * 2 + 1];
    float dx = pos[dst * 2 + 0] - pos[src * 2 + 0];
    float dy = pos[dst * 2 + 1] - pos[src * 2 + 1];
    float dist = __builtin_amdgcn_sqrtf(dx * dx + dy * dy);  // raw v_sqrt_f32

    // B fragments from LDS (ds_load_b128 pairs)
    const _Float16* fb = &ldsWm[(size_t)lane * 16];
    v16h b00 = cat8(*(const v8h*)(fb + 0 * 512), *(const v8h*)(fb + 0 * 512 + 8));
    v16h b01 = cat8(*(const v8h*)(fb + 1 * 512), *(const v8h*)(fb + 1 * 512 + 8));
    v16h b10 = cat8(*(const v8h*)(fb + 2 * 512), *(const v8h*)(fb + 2 * 512 + 8));
    v16h b11 = cat8(*(const v8h*)(fb + 3 * 512), *(const v8h*)(fb + 3 * 512 + 8));

    // A fragments straight from f16 mirror (b128 pairs, no cvt)
    int ka = half * 8;
    const _Float16* ri = h16 + (size_t)dst * HH;   // x_i
    const _Float16* rj = h16 + (size_t)src * HH;   // x_j
    v16h a0 = cat8(*(const v8h*)(ri + ka), *(const v8h*)(ri + 16 + ka));
    v16h a1 = cat8(*(const v8h*)(rj + ka), *(const v8h*)(rj + 16 + ka));

    v8f c0 = {}; v8f c1 = {};
    c0 = wmma_f16(a0, b00, c0);
    c1 = wmma_f16(a0, b01, c1);
    c0 = wmma_f16(a1, b10, c0);
    c1 = wmma_f16(a1, b11, c1);

    // epilogue: + dist * Wm[64,:] + bm, SiLU, atomic scatter
    int n = m;
    float w640 = Wm[64 * HH + n];
    float w641 = Wm[64 * HH + n + 16];
    float bm0 = bm[n], bm1 = bm[n + 16];
#pragma unroll
    for (int r = 0; r < 8; ++r) {
        int M = r + 8 * half;
        float dM  = __shfl(dist, M, 32);
        int  dstM = __shfl(dst,  M, 32);
        float p0 = silu(c0[r] + dM * w640 + bm0);
        float p1 = silu(c1[r] + dM * w641 + bm1);
        unsafeAtomicAdd(&aggr[(size_t)dstM * HH + n],      p0);
        unsafeAtomicAdd(&aggr[(size_t)dstM * HH + n + 16], p1);
    }
}

// ---------------------------------------------------------------------------
// h = h + silu([h, aggr] @ Wu + bu)   (in place; wave owns 16 consecutive nodes)
// ---------------------------------------------------------------------------
__global__ void update_kernel(float* h, _Float16* h16,
                              const float* __restrict__ aggr,
                              const _Float16* __restrict__ wuF,
                              const float* __restrict__ bu) {
    int wave = (blockIdx.x * blockDim.x + threadIdx.x) >> 5;
    int lane = threadIdx.x & 31;
    int n0 = wave << 4;
    if (n0 >= NN) return;                 // wave-uniform (NN % 16 == 0)
    int half = lane >> 4;
    int m = lane & 15;
    int node = n0 + m;

    const _Float16* fb = wuF + (size_t)lane * 16;
    v16h b00 = cat8(*(const v8h*)(fb + 0 * 512), *(const v8h*)(fb + 0 * 512 + 8));
    v16h b01 = cat8(*(const v8h*)(fb + 1 * 512), *(const v8h*)(fb + 1 * 512 + 8));
    v16h b10 = cat8(*(const v8h*)(fb + 2 * 512), *(const v8h*)(fb + 2 * 512 + 8));
    v16h b11 = cat8(*(const v8h*)(fb + 3 * 512), *(const v8h*)(fb + 3 * 512 + 8));

    int ka = half * 8;
    const _Float16* rh = h16 + (size_t)node * HH;
    v16h a0 = cat8(*(const v8h*)(rh + ka), *(const v8h*)(rh + 16 + ka));

    const float* ra = aggr + (size_t)node * HH;
    v16h a1;
#pragma unroll
    for (int i = 0; i < 8; ++i) {
        a1[i]     = (_Float16)ra[ka + i];
        a1[8 + i] = (_Float16)ra[16 + ka + i];
    }

    v8f c0 = {}; v8f c1 = {};
    c0 = wmma_f16(a0, b00, c0);
    c1 = wmma_f16(a0, b01, c1);
    c0 = wmma_f16(a1, b10, c0);
    c1 = wmma_f16(a1, b11, c1);

    int n = m;
    float bu0 = bu[n], bu1 = bu[n + 16];
    float res0[8], res1[8];
#pragma unroll
    for (int r = 0; r < 8; ++r) {
        int M = r + 8 * half;
        res0[r] = h[(size_t)(n0 + M) * HH + n];
        res1[r] = h[(size_t)(n0 + M) * HH + n + 16];
    }
#pragma unroll
    for (int r = 0; r < 8; ++r) {
        int M = r + 8 * half;
        float o0 = res0[r] + silu(c0[r] + bu0);
        float o1 = res1[r] + silu(c1[r] + bu1);
        h[(size_t)(n0 + M) * HH + n]        = o0;
        h[(size_t)(n0 + M) * HH + n + 16]   = o1;
        h16[(size_t)(n0 + M) * HH + n]      = (_Float16)o0;
        h16[(size_t)(n0 + M) * HH + n + 16] = (_Float16)o1;
    }
}

// ---------------------------------------------------------------------------
// nbr[dst] += h[src]   (thread per (edge, feature))
// ---------------------------------------------------------------------------
__global__ void nbr_scatter_kernel(const float* __restrict__ h,
                                   const int* __restrict__ edge,
                                   float* __restrict__ nbr) {
    long long t = (long long)blockIdx.x * blockDim.x + threadIdx.x;
    if (t >= (long long)EE * HH) return;
    int e = (int)(t >> 5);
    int f = (int)(t & 31);
    int src = edge[e * 2 + 0];
    int dst = edge[e * 2 + 1];
    unsafeAtomicAdd(&nbr[(size_t)dst * HH + f], h[(size_t)src * HH + f]);
}

// ---------------------------------------------------------------------------
// score = nbr @ Wr + br + h @ Wl
// ---------------------------------------------------------------------------
__global__ void score_kernel(const float* __restrict__ h,
                             const float* __restrict__ nbr,
                             const float* __restrict__ Wr,
                             const float* __restrict__ br,
                             const float* __restrict__ Wl,
                             float* __restrict__ score) {
    int n = blockIdx.x * blockDim.x + threadIdx.x;
    if (n >= NN) return;
    float acc = br[0];
#pragma unroll
    for (int i = 0; i < HH; ++i)
        acc += nbr[(size_t)n * HH + i] * Wr[i] + h[(size_t)n * HH + i] * Wl[i];
    score[n] = acc;
}

// ---------------------------------------------------------------------------
// Top-K keys: (~orderable(score) << 32) | idx ; ascending sort == score desc.
// ---------------------------------------------------------------------------
__global__ void build_keys_kernel(const float* __restrict__ score,
                                  u64* __restrict__ keys) {
    int i = blockIdx.x * blockDim.x + threadIdx.x;
    if (i >= SS) return;
    if (i < NN) {
        unsigned u = __float_as_uint(score[i]);
        u = (u & 0x80000000u) ? ~u : (u | 0x80000000u);
        keys[i] = ((u64)(~u) << 32) | (unsigned)i;
    } else {
        keys[i] = 0xFFFFFFFFFFFFFFFFull;
    }
}

// LDS-staged bitonic pre-sort: all stages k=2..2048 inside one block (16 KB LDS)
__global__ void bitonic_presort_kernel(u64* __restrict__ d) {
    __shared__ u64 s[2048];
    int base = blockIdx.x * 2048;
    int t = threadIdx.x;                 // 256 threads
    for (int i = t; i < 2048; i += 256) s[i] = d[base + i];
    __syncthreads();
    for (int k = 2; k <= 2048; k <<= 1) {
        for (int j = k >> 1; j > 0; j >>= 1) {
            for (int p = t; p < 1024; p += 256) {
                int i = ((p & ~(j - 1)) << 1) | (p & (j - 1));
                int ixj = i | j;
                bool up = (((base + i) & k) == 0);
                u64 a = s[i], b = s[ixj];
                if ((a > b) == up) { s[i] = b; s[ixj] = a; }
            }
            __syncthreads();
        }
    }
    for (int i = t; i < 2048; i += 256) d[base + i] = s[i];
}

// global compare-exchange step (j >= 2048)
__global__ void bitonic_step_kernel(u64* __restrict__ d, int j, int k) {
    int i = blockIdx.x * blockDim.x + threadIdx.x;
    int ixj = i ^ j;
    if (ixj > i) {
        u64 a = d[i], b = d[ixj];
        bool up = ((i & k) == 0);
        if ((a > b) == up) { d[i] = b; d[ixj] = a; }
    }
}

// LDS-staged merge tail: j = 1024..1 for a given k (direction uniform per block)
__global__ void bitonic_merge_kernel(u64* __restrict__ d, int k) {
    __shared__ u64 s[2048];
    int base = blockIdx.x * 2048;
    int t = threadIdx.x;
    for (int i = t; i < 2048; i += 256) s[i] = d[base + i];
    __syncthreads();
    bool up = ((base & k) == 0);         // k >= 4096 > block span -> uniform
    for (int j = 1024; j > 0; j >>= 1) {
        for (int p = t; p < 1024; p += 256) {
            int i = ((p & ~(j - 1)) << 1) | (p & (j - 1));
            int ixj = i | j;
            u64 a = s[i], b = s[ixj];
            if ((a > b) == up) { s[i] = b; s[ixj] = a; }
        }
        __syncthreads();
    }
    for (int i = t; i < 2048; i += 256) d[base + i] = s[i];
}

// ---------------------------------------------------------------------------
// x_pool = (h[perm] * tanh(score[perm])) @ Wo + bo ; batch ids after K*32 floats
// ---------------------------------------------------------------------------
__global__ void pool_kernel(const _Float16* __restrict__ h16,
                            const float* __restrict__ score,
                            const u64*   __restrict__ keys,
                            const _Float16* __restrict__ woF,
                            const float* __restrict__ bo,
                            const int*   __restrict__ batch,
                            float* __restrict__ out) {
    int wave = (blockIdx.x * blockDim.x + threadIdx.x) >> 5;
    int lane = threadIdx.x & 31;
    int k0 = wave << 4;
    if (k0 >= KK) return;                 // wave-uniform
    int half = lane >> 4;
    int m = lane & 15;
    int km = k0 + m; if (km >= KK) km = KK - 1;   // clamp reads; guard writes
    int idx = (int)(keys[km] & 0xFFFFFFFFull);
    _Float16 scale = (_Float16)tanhf(score[idx]);

    int ka = half * 8;
    const _Float16* r = h16 + (size_t)idx * HH;
    v8h lo = *(const v8h*)(r + ka);
    v8h hi = *(const v8h*)(r + 16 + ka);
    v16h a;
#pragma unroll
    for (int i = 0; i < 8; ++i) { a[i] = lo[i] * scale; a[8 + i] = hi[i] * scale; }

    const _Float16* fb = woF + (size_t)lane * 16;
    v16h b0 = cat8(*(const v8h*)(fb + 0 * 512), *(const v8h*)(fb + 0 * 512 + 8));
    v16h b1 = cat8(*(const v8h*)(fb + 1 * 512), *(const v8h*)(fb + 1 * 512 + 8));

    v8f c0 = {}; v8f c1 = {};
    c0 = wmma_f16(a, b0, c0);
    c1 = wmma_f16(a, b1, c1);

    int n = m;
    float bo0 = bo[n], bo1 = bo[n + 16];
#pragma unroll
    for (int r8 = 0; r8 < 8; ++r8) {
        int M = r8 + 8 * half;
        int kr = k0 + M;
        if (kr < KK) {
            out[(size_t)kr * HH + n]      = c0[r8] + bo0;
            out[(size_t)kr * HH + n + 16] = c1[r8] + bo1;
        }
    }
    if (half == 0 && (k0 + m) < KK) {
        int id2 = (int)(keys[k0 + m] & 0xFFFFFFFFull);
        ((int*)(out + (size_t)KK * HH))[k0 + m] = batch[id2];
    }
}

// ---------------------------------------------------------------------------
extern "C" void kernel_launch(void* const* d_in, const int* in_sizes, int n_in,
                              void* d_out, int out_size, void* d_ws, size_t ws_size,
                              hipStream_t stream) {
    const float* x    = (const float*)d_in[0];
    const float* pos  = (const float*)d_in[1];
    const int*   edge = (const int*)d_in[2];
    const int*   batch= (const int*)d_in[3];
    const float* We = (const float*)d_in[4];
    const float* be = (const float*)d_in[5];
    const float* Wm = (const float*)d_in[6];
    const float* bm = (const float*)d_in[7];
    const float* Wu = (const float*)d_in[8];
    const float* bu = (const float*)d_in[9];
    const float* Wr = (const float*)d_in[10];
    const float* br = (const float*)d_in[11];
    const float* Wl = (const float*)d_in[12];
    const float* Wo = (const float*)d_in[13];
    const float* bo = (const float*)d_in[14];
    float* out = (float*)d_out;

    char* ws = (char*)d_ws;
    float*     h     = (float*)(ws);                       // 12.8 MB
    float*     aggr  = (float*)(ws + (size_t)12800000);    // 12.8 MB (reused as nbr)
    float*     score = (float*)(ws + (size_t)25600000);    // 0.4 MB
    u64*       keys  = (u64*)  (ws + (size_t)26000000);    // 1 MB
    _Float16*  h16   = (_Float16*)(ws + (size_t)27048576); // 6.4 MB (16B aligned)
    _Float16*  wmF   = (_Float16*)(ws + (size_t)33448576); // 4 KB
    _Float16*  wuF   = (_Float16*)(ws + (size_t)33452672); // 4 KB
    _Float16*  woF   = (_Float16*)(ws + (size_t)33456768); // 2 KB

    // 0. pre-pack f16 WMMA B-fragments
    prep_frags_kernel<<<1, 128, 0, stream>>>(Wm, Wu, Wo, wmF, wuF, woF);

    // 1. embed (f32 + f16 mirror)
    embed_kernel<<<(NN * HH + 255) / 256, 256, 0, stream>>>(x, pos, We, be, h, h16);

    // 2. edge messages -> aggr  (TDM-staged B fragments, 4 WMMA / 16 edges)
    zero_f32<<<(NN * HH + 255) / 256, 256, 0, stream>>>(aggr, NN * HH);
    edge_msg_kernel<<<(EE / 16) / 8, 256, 0, stream>>>(h16, pos, edge, wmF, Wm, bm, aggr);

    // 3. node update (in place, refresh f16 mirror)
    update_kernel<<<((NN / 16) + 7) / 8, 256, 0, stream>>>(h, h16, aggr, wuF, bu);

    // 4. nbr segment-sum (reuse aggr buffer)
    zero_f32<<<(NN * HH + 255) / 256, 256, 0, stream>>>(aggr, NN * HH);
    {
        long long tot = (long long)EE * HH;
        nbr_scatter_kernel<<<(unsigned)((tot + 255) / 256), 256, 0, stream>>>(h, edge, aggr);
    }

    // 5. score
    score_kernel<<<(NN + 255) / 256, 256, 0, stream>>>(h, aggr, Wr, br, Wl, score);

    // 6. exact top-K: LDS-staged bitonic sort (28 launches)
    build_keys_kernel<<<SS / 256, 256, 0, stream>>>(score, keys);
    bitonic_presort_kernel<<<SS / 2048, 256, 0, stream>>>(keys);
    for (int k = 4096; k <= SS; k <<= 1) {
        for (int j = k >> 1; j >= 2048; j >>= 1)
            bitonic_step_kernel<<<SS / 256, 256, 0, stream>>>(keys, j, k);
        bitonic_merge_kernel<<<SS / 2048, 256, 0, stream>>>(keys, k);
    }

    // 7. pool + output GEMM
    {
        int waves = (KK + 15) / 16;
        pool_kernel<<<(waves + 7) / 8, 256, 0, stream>>>(h16, score, keys, woF, bo, batch, out);
    }
}
